// RotaryEmbeddingSelfAttention_49761491092098
// MI455X (gfx1250) — matrix operand order repaired and checked by reading, not compiled, not run on hardware
//
#include <hip/hip_runtime.h>
#include <hip/hip_bf16.h>

typedef __bf16 bf16;
typedef __attribute__((ext_vector_type(16))) __bf16 v16bf;
typedef __attribute__((ext_vector_type(8)))  __bf16 v8bf;
typedef __attribute__((ext_vector_type(8)))  float  v8f;

constexpr int B_ = 2, S_ = 2048, D_ = 1024, H_ = 16, HD_ = 64;
constexpr int M_ = B_ * S_;   // 4096 rows

union ABCast { v16bf v; v8bf h[2]; };

// A-fragment (16x32 bf16): lanes 0-15 row M=lane, K {k0..k0+7, k0+16..k0+23};
// lanes 16-31 same rows, K {k0+8..k0+15, k0+24..k0+31}.  (ISA 7.12.2)
__device__ inline v16bf load_frag_a(const bf16* base, int row0, int ld, int k0, int lane) {
  int r  = lane & 15;
  int kk = k0 + ((lane & 16) >> 1);   // +8 elements for upper half-wave
  const bf16* p = base + (size_t)(row0 + r) * ld + kk;
  ABCast f;
  f.h[0] = *(const v8bf*)(p);
  f.h[1] = *(const v8bf*)(p + 16);
  return f.v;
}

// B-fragment (32x16 bf16), memory holds B^T (N rows, K contiguous):
// lanes 0-15 col N=lane, K {k0..k0+7, k0+16..k0+23}; lanes 16-31 K {k0+8..k0+15, k0+24..k0+31}.
__device__ inline v16bf load_frag_b(const bf16* base, int row0, int ld, int k0, int lane) {
  int r  = lane & 15;
  int kk = k0 + ((lane & 16) >> 1);   // +8 elements for upper half-wave
  const bf16* p = base + (size_t)(row0 + r) * ld + kk;
  ABCast f;
  f.h[0] = *(const v8bf*)(p);
  f.h[1] = *(const v8bf*)(p + 16);
  return f.v;
}

__device__ inline v8f wmma_bf16(v16bf a, v16bf b, v8f c) {
  return __builtin_amdgcn_wmma_f32_16x16x32_bf16(false, a, false, b, (short)0, c, false, false);
}

__device__ inline v8f vzero8() { v8f z = {0.f,0.f,0.f,0.f,0.f,0.f,0.f,0.f}; return z; }

// ---------------------------------------------------------------- converts
__global__ void convert_kernel(const float* __restrict__ in, bf16* __restrict__ out, int n) {
  int i = blockIdx.x * 256 + threadIdx.x;
  if (i < n) out[i] = (bf16)in[i];
}

// out[n*D + k] = in[k*D + n]  (weight transpose + bf16 convert)
__global__ void transpose_conv_kernel(const float* __restrict__ in, bf16* __restrict__ out) {
  int i = blockIdx.x * 256 + threadIdx.x;       // over D*D
  int n = i >> 10, k = i & (D_ - 1);
  out[i] = (bf16)in[(size_t)k * D_ + n];
}

// ---------------------------------------------------------------- GEMM epilogues
// MODE 0: f32 out [M,D].  MODE 1: RoPE -> bf16 [B,H,S,HD].  MODE 2: bf16 [B,H,HD,S].
template <int MODE>
__device__ inline void store_tile(v8f (&acc)[4], int m0, int n0, int lane,
                                  const float* __restrict__ bias, void* __restrict__ outv) {
  const int rowhi = (lane >> 4) << 3;   // C layout: vgpr r -> row r (+8 upper half)
  const int nl    = lane & 15;

  #pragma unroll
  for (int j = 0; j < 4; ++j) {
    float bc = bias[n0 + j * 16 + nl];
    #pragma unroll
    for (int r = 0; r < 8; ++r) acc[j][r] += bc;
  }

  if (MODE == 0) {
    float* out = (float*)outv;
    #pragma unroll
    for (int j = 0; j < 4; ++j)
      #pragma unroll
      for (int r = 0; r < 8; ++r)
        out[(size_t)(m0 + r + rowhi) * D_ + n0 + j * 16 + nl] = acc[j][r];
  } else if (MODE == 1) {
    // RoPE: pairs (d, d+32) of one head live in acc[jp] / acc[jp+2], same lane.
    bf16* out = (bf16*)outv;
    #pragma unroll
    for (int jp = 0; jp < 2; ++jp) {
      int   ncol = n0 + jp * 16 + nl;
      int   h    = ncol >> 6;
      int   dlo  = ncol & 63;                           // in [0,32)
      float invf = exp2f((float)dlo * -0.4152410118f);  // 10000^(-2d/64)
      #pragma unroll
      for (int r = 0; r < 8; ++r) {
        int   mg  = m0 + r + rowhi;
        int   s   = mg & (S_ - 1);
        int   b   = mg >> 11;
        float th  = (float)s * invf;
        float cth = __cosf(th), sth = __sinf(th);
        float x1 = acc[jp][r], x2 = acc[jp + 2][r];
        size_t basei = (((size_t)b * H_ + h) * S_ + s) * HD_;
        out[basei + dlo]      = (bf16)(x1 * cth - x2 * sth);
        out[basei + dlo + 32] = (bf16)(x2 * cth + x1 * sth);
      }
    }
  } else {  // MODE 2: V transposed [B,H,HD,S]
    bf16* out = (bf16*)outv;
    #pragma unroll
    for (int j = 0; j < 4; ++j)
      #pragma unroll
      for (int r = 0; r < 8; ++r) {
        int mg = m0 + r + rowhi;
        int s  = mg & (S_ - 1);
        int b  = mg >> 11;
        int nc = n0 + j * 16 + nl;
        int h  = nc >> 6;
        int d  = nc & 63;
        out[(((size_t)b * H_ + h) * HD_ + d) * S_ + s] = (bf16)acc[j][r];
      }
  }
}

// C[m,n] = sum_k A[m,k] * W[k,n] + bias[n], A bf16 [M,D], Bt = W^T bf16 [D,D].
// Wave tile 32(M) x 64(N).  Two-phase register ping-pong: each phase's 12 b128
// loads issue before the other phase's 8 WMMAs (no copy chains).
template <int MODE>
__global__ __launch_bounds__(256)
void gemm_kernel(const bf16* __restrict__ A, const bf16* __restrict__ Bt,
                 const float* __restrict__ bias, void* __restrict__ outv) {
  const int lane = threadIdx.x & 31;
  const int wid  = threadIdx.x >> 5;
  const int m0 = blockIdx.y * 64  + (wid >> 2) * 32;
  const int n0 = blockIdx.x * 256 + (wid & 3) * 64;   // 64-wide = one head-aligned span

  v8f acc[2][4];
  #pragma unroll
  for (int i = 0; i < 2; ++i)
    #pragma unroll
    for (int j = 0; j < 4; ++j) acc[i][j] = vzero8();

  // prologue: phase-0 fragments (k = 0)
  v16bf a0 = load_frag_a(A, m0,      D_, 0, lane);
  v16bf a1 = load_frag_a(A, m0 + 16, D_, 0, lane);
  v16bf b0[4];
  #pragma unroll
  for (int j = 0; j < 4; ++j) b0[j] = load_frag_b(Bt, n0 + j * 16, D_, 0, lane);

  for (int k = 0; k < D_; k += 64) {
    __builtin_prefetch(A  + (size_t)(m0 + lane) * D_ + k + 96, 0, 3);
    __builtin_prefetch(Bt + (size_t)(n0 + lane) * D_ + k + 96, 0, 3);
    __builtin_prefetch(Bt + (size_t)(n0 + 32 + lane) * D_ + k + 96, 0, 3);

    // phase 1 loads (k+32) issue ahead of phase 0 WMMAs
    v16bf c0 = load_frag_a(A, m0,      D_, k + 32, lane);
    v16bf c1 = load_frag_a(A, m0 + 16, D_, k + 32, lane);
    v16bf b1[4];
    #pragma unroll
    for (int j = 0; j < 4; ++j) b1[j] = load_frag_b(Bt, n0 + j * 16, D_, k + 32, lane);

    #pragma unroll
    for (int j = 0; j < 4; ++j) {
      acc[0][j] = wmma_bf16(a0, b0[j], acc[0][j]);
      acc[1][j] = wmma_bf16(a1, b0[j], acc[1][j]);
    }

    // phase 0 loads for next outer iteration (k+64) ahead of phase 1 WMMAs
    if (k + 64 < D_) {
      a0 = load_frag_a(A, m0,      D_, k + 64, lane);
      a1 = load_frag_a(A, m0 + 16, D_, k + 64, lane);
      #pragma unroll
      for (int j = 0; j < 4; ++j) b0[j] = load_frag_b(Bt, n0 + j * 16, D_, k + 64, lane);
    }

    #pragma unroll
    for (int j = 0; j < 4; ++j) {
      acc[0][j] = wmma_bf16(c0, b1[j], acc[0][j]);
      acc[1][j] = wmma_bf16(c1, b1[j], acc[1][j]);
    }
  }

  store_tile<MODE>(acc[0], m0,      n0, lane, bias, outv);
  store_tile<MODE>(acc[1], m0 + 16, n0, lane, bias, outv);
}

// ---------------------------------------------------------------- attention
// Q,K: bf16 [B,H,S,HD]; Vt: bf16 [B,H,HD,S]; Out: bf16 [B*S, D] (= [b,s,h*64+d])
__global__ __launch_bounds__(128)
void attn_kernel(const bf16* __restrict__ Q, const bf16* __restrict__ K,
                 const bf16* __restrict__ Vt, bf16* __restrict__ Out) {
  __shared__ alignas(16) bf16 psh[4][16][32];   // per-wave P staging tile
  const int lane = threadIdx.x & 31;
  const int wid  = threadIdx.x >> 5;
  const int wt   = blockIdx.x * 4 + wid;
  const int ST   = S_ / 16;
  const int qt = wt % ST;
  const int h  = (wt / ST) % H_;
  const int b  =  wt / (ST * H_);
  const int qb = qt * 16;

  const bf16* Qh = Q  + (((size_t)b * H_ + h) * S_)  * HD_;
  const bf16* Kh = K  + (((size_t)b * H_ + h) * S_)  * HD_;
  const bf16* Vh = Vt + (((size_t)b * H_ + h) * HD_) * S_;

  const v16bf aq0 = load_frag_a(Qh, qb, HD_, 0,  lane);
  const v16bf aq1 = load_frag_a(Qh, qb, HD_, 32, lane);

  v8f acc[4];
  #pragma unroll
  for (int j = 0; j < 4; ++j) acc[j] = vzero8();
  float mrow[8], lrow[8];
  #pragma unroll
  for (int r = 0; r < 8; ++r) { mrow[r] = -1e30f; lrow[r] = 0.f; }

  const int rowhi = (lane >> 4) << 3;
  const int nl    = lane & 15;

  // prologue: K-fragments for first 32-key chunk
  v16bf bk[4];
  bk[0] = load_frag_b(Kh, 0,  HD_, 0,  lane);
  bk[1] = load_frag_b(Kh, 0,  HD_, 32, lane);
  bk[2] = load_frag_b(Kh, 16, HD_, 0,  lane);
  bk[3] = load_frag_b(Kh, 16, HD_, 32, lane);

  for (int kb = 0; kb < qb + 16; kb += 32) {
    v8f s0 = vzero8(), s1 = vzero8();
    s0 = wmma_bf16(aq0, bk[0], s0);
    s0 = wmma_bf16(aq1, bk[1], s0);
    s1 = wmma_bf16(aq0, bk[2], s1);
    s1 = wmma_bf16(aq1, bk[3], s1);

    // V fragments for this chunk + K fragments for the next chunk:
    // their latency hides under the softmax VALU section below.
    v16bf bv[4];
    #pragma unroll
    for (int j = 0; j < 4; ++j) bv[j] = load_frag_b(Vh, j * 16, S_, kb, lane);
    if (kb + 32 < qb + 16) {
      bk[0] = load_frag_b(Kh, kb + 32, HD_, 0,  lane);
      bk[1] = load_frag_b(Kh, kb + 32, HD_, 32, lane);
      bk[2] = load_frag_b(Kh, kb + 48, HD_, 0,  lane);
      bk[3] = load_frag_b(Kh, kb + 48, HD_, 32, lane);
    }

    #pragma unroll
    for (int r = 0; r < 8; ++r) { s0[r] *= 0.125f; s1[r] *= 0.125f; }

    if (kb + 32 > qb) {            // chunk straddles the causal diagonal
      #pragma unroll
      for (int r = 0; r < 8; ++r) {
        int qrow = qb + r + rowhi;
        if (kb + nl      > qrow) s0[r] = -1e30f;
        if (kb + 16 + nl > qrow) s1[r] = -1e30f;
      }
    }

    float scale[8];
    #pragma unroll
    for (int r = 0; r < 8; ++r) {
      float mx = fmaxf(s0[r], s1[r]);
      mx = fmaxf(mx, __shfl_xor(mx, 1, 32));
      mx = fmaxf(mx, __shfl_xor(mx, 2, 32));
      mx = fmaxf(mx, __shfl_xor(mx, 4, 32));
      mx = fmaxf(mx, __shfl_xor(mx, 8, 32));   // reduce within 16-lane half
      float nm = fmaxf(mrow[r], mx);
      scale[r] = __expf(mrow[r] - nm);
      mrow[r]  = nm;
      s0[r] = __expf(s0[r] - nm);
      s1[r] = __expf(s1[r] - nm);
      float rs = s0[r] + s1[r];
      rs += __shfl_xor(rs, 1, 32);
      rs += __shfl_xor(rs, 2, 32);
      rs += __shfl_xor(rs, 4, 32);
      rs += __shfl_xor(rs, 8, 32);
      lrow[r] = lrow[r] * scale[r] + rs;
    }
    #pragma unroll
    for (int j = 0; j < 4; ++j)
      #pragma unroll
      for (int r = 0; r < 8; ++r) acc[j][r] *= scale[r];

    // C-layout -> A-fragment via wave-private LDS tile (DS ops in-order per wave)
    #pragma unroll
    for (int r = 0; r < 8; ++r) {
      psh[wid][r + rowhi][nl]      = (bf16)s0[r];
      psh[wid][r + rowhi][nl + 16] = (bf16)s1[r];
    }
    asm volatile("" ::: "memory");
    v16bf ap = load_frag_a(&psh[wid][0][0], 0, 32, 0, lane);
    asm volatile("" ::: "memory");

    #pragma unroll
    for (int j = 0; j < 4; ++j) acc[j] = wmma_bf16(ap, bv[j], acc[j]);
  }

  #pragma unroll
  for (int r = 0; r < 8; ++r) {
    float inv = 1.0f / lrow[r];
    int   s   = qb + r + rowhi;
    size_t ob = ((size_t)b * S_ + s) * D_ + h * HD_;
    #pragma unroll
    for (int j = 0; j < 4; ++j)
      Out[ob + j * 16 + nl] = (bf16)(acc[j][r] * inv);
  }
}

// ---------------------------------------------------------------- launch
extern "C" void kernel_launch(void* const* d_in, const int* in_sizes, int n_in,
                              void* d_out, int out_size, void* d_ws, size_t ws_size,
                              hipStream_t stream) {
  (void)in_sizes; (void)n_in; (void)out_size; (void)ws_size;
  const float* x  = (const float*)d_in[0];
  // d_in[1] = boolean causal mask: applied analytically, not read.
  const float* Wq = (const float*)d_in[2];
  const float* bq = (const float*)d_in[3];
  const float* Wk = (const float*)d_in[4];
  const float* bk = (const float*)d_in[5];
  const float* Wv = (const float*)d_in[6];
  const float* bv = (const float*)d_in[7];
  const float* Wo = (const float*)d_in[8];
  const float* bo = (const float*)d_in[9];

  char* ws = (char*)d_ws;
  const size_t MD = (size_t)M_ * D_ * sizeof(bf16);   // 8 MB
  const size_t DD = (size_t)D_ * D_ * sizeof(bf16);   // 2 MB
  bf16* Xbf = (bf16*)ws; ws += MD;
  bf16* Wtq = (bf16*)ws; ws += DD;
  bf16* Wtk = (bf16*)ws; ws += DD;
  bf16* Wtv = (bf16*)ws; ws += DD;
  bf16* Wto = (bf16*)ws; ws += DD;
  bf16* Qbf = (bf16*)ws; ws += MD;
  bf16* Kbf = (bf16*)ws; ws += MD;
  bf16* Vtb = (bf16*)ws; ws += MD;
  bf16* Abf = (bf16*)ws; ws += MD;   // total ~48 MB

  convert_kernel<<<(M_ * D_) / 256, 256, 0, stream>>>(x, Xbf, M_ * D_);
  transpose_conv_kernel<<<(D_ * D_) / 256, 256, 0, stream>>>(Wq, Wtq);
  transpose_conv_kernel<<<(D_ * D_) / 256, 256, 0, stream>>>(Wk, Wtk);
  transpose_conv_kernel<<<(D_ * D_) / 256, 256, 0, stream>>>(Wv, Wtv);
  transpose_conv_kernel<<<(D_ * D_) / 256, 256, 0, stream>>>(Wo, Wto);

  dim3 ggrid(D_ / 256, M_ / 64);
  gemm_kernel<1><<<ggrid, 256, 0, stream>>>(Xbf, Wtq, bq, Qbf);
  gemm_kernel<1><<<ggrid, 256, 0, stream>>>(Xbf, Wtk, bk, Kbf);
  gemm_kernel<2><<<ggrid, 256, 0, stream>>>(Xbf, Wtv, bv, Vtb);

  attn_kernel<<<(B_ * H_ * (S_ / 16)) / 4, 128, 0, stream>>>(Qbf, Kbf, Vtb, Abf);

  gemm_kernel<0><<<ggrid, 256, 0, stream>>>(Abf, Wto, bo, d_out);
}
